// Attention_6382321402310
// MI455X (gfx1250) — matrix operand order repaired and checked by reading, not compiled, not run on hardware
//
#include <hip/hip_runtime.h>
#include <hip/hip_bf16.h>

// Problem constants (match reference)
#define B_   2
#define L_   2048
#define DIM_ 1024
#define H_   16
#define HD_  64
#define TDIM (3 * DIM_)

typedef __bf16 bf16_t;
typedef __attribute__((ext_vector_type(16))) __bf16 v16bf;
typedef __attribute__((ext_vector_type(8)))  __bf16 v8bf;
typedef __attribute__((ext_vector_type(8)))  float  v8f;

// Build a 16-element bf16 fragment from two contiguous 8-element (16B) chunks.
__device__ __forceinline__ v16bf make_frag(const bf16_t* lo, const bf16_t* hi) {
    v8bf a = *(const v8bf*)lo;
    v8bf b = *(const v8bf*)hi;
    v16bf r;
#pragma unroll
    for (int i = 0; i < 8; ++i) { r[i] = a[i]; r[i + 8] = b[i]; }
    return r;
}

// CDNA5 async copy: global -> LDS, 16 bytes per lane, tracked by ASYNCcnt.
__device__ __forceinline__ void async_load16(unsigned lds_off, const bf16_t* gptr) {
    asm volatile("global_load_async_to_lds_b128 %0, %1, off"
                 :: "v"(lds_off), "v"(gptr) : "memory");
}
__device__ __forceinline__ void wait_async0() {
    asm volatile("s_wait_asynccnt 0x0" ::: "memory");
}

// ---------------- conversion kernels ----------------

__global__ void k_cvt_f32_bf16(const float* __restrict__ in, bf16_t* __restrict__ out, int n) {
    int i = blockIdx.x * blockDim.x + threadIdx.x;
    if (i < n) out[i] = (bf16_t)in[i];
}

// in: [K][N] f32 row-major  ->  out: [N][K] bf16 (transposed)
__global__ void k_transpose_cvt(const float* __restrict__ in, bf16_t* __restrict__ out,
                                int K, int N) {
    int i = blockIdx.x * blockDim.x + threadIdx.x;
    if (i < K * N) {
        int n = i / K;
        int k = i - n * K;
        out[i] = (bf16_t)in[(size_t)k * N + n];
    }
}

// ---------------- bf16 WMMA GEMM: C[M,N] = A[M,K] @ Bt[N,K]^T (+bias) ----------------
// One wave computes a 32(M) x 64(N) tile. The k-loop is unrolled 2x into a
// ping-pong over two fragment sets: no register copies, loads of one set overlap
// the WMMAs consuming the other.

__global__ void k_gemm_wmma(const bf16_t* __restrict__ A, const bf16_t* __restrict__ Bt,
                            float* __restrict__ C, const float* __restrict__ bias,
                            int M, int N, int K) {
    const int lane = threadIdx.x;      // 0..31
    const int half = lane >> 4;        // 0 or 1
    const int l16  = lane & 15;
    const int m0 = blockIdx.x * 32;
    const int n0 = blockIdx.y * 64;

    v8f acc[2][4];
#pragma unroll
    for (int mi = 0; mi < 2; ++mi)
#pragma unroll
        for (int t = 0; t < 4; ++t) acc[mi][t] = (v8f){};

    const bf16_t* arow0 = A + (size_t)(m0 + l16) * K;
    const bf16_t* arow1 = A + (size_t)(m0 + 16 + l16) * K;
    const bf16_t* brow  = Bt + (size_t)(n0 + l16) * K;

    v16bf a[2][2], b[2][4];   // [set][...]

    auto load_set = [&](int s, int k0) {
        a[s][0] = make_frag(arow0 + k0 + 8 * half, arow0 + k0 + 16 + 8 * half);
        a[s][1] = make_frag(arow1 + k0 + 8 * half, arow1 + k0 + 16 + 8 * half);
#pragma unroll
        for (int t = 0; t < 4; ++t) {
            const bf16_t* p = brow + (size_t)(16 * t) * K + k0 + 8 * half;
            b[s][t] = make_frag(p, p + 16);
        }
    };
    auto mma_set = [&](int s) {
#pragma unroll
        for (int t = 0; t < 4; ++t) {
            acc[0][t] = __builtin_amdgcn_wmma_f32_16x16x32_bf16(
                false, a[s][0], false, b[s][t], (short)0, acc[0][t], false, false);
            acc[1][t] = __builtin_amdgcn_wmma_f32_16x16x32_bf16(
                false, a[s][1], false, b[s][t], (short)0, acc[1][t], false, false);
        }
    };

    load_set(0, 0);
    int k0 = 0;
    for (; k0 < K - 64; k0 += 64) {
        load_set(1, k0 + 32);
        mma_set(0);
        load_set(0, k0 + 64);
        mma_set(1);
    }
    // final 64: set0 holds K-64 fragments
    load_set(1, K - 32);
    mma_set(0);
    mma_set(1);

#pragma unroll
    for (int mi = 0; mi < 2; ++mi)
#pragma unroll
        for (int t = 0; t < 4; ++t)
#pragma unroll
            for (int v = 0; v < 8; ++v) {
                int row = m0 + 16 * mi + v + 8 * half;
                int col = n0 + 16 * t + l16;
                float r = acc[mi][t][v];
                if (bias) r += bias[col];
                C[(size_t)row * N + col] = r;
            }
}

// ---------------- fused RMSNorm + RoPE + layout change ----------------
// One wave handles one (b, l, h) row of 64 elements. Lane i owns elements i and i+32
// (exactly the RoPE rotate-half pairing).
// Outputs: Qb/Kb [b*H+h][l][hd] bf16,  Vt [b*H+h][hd][l] bf16 (transposed for P@V).

__global__ void k_norm_rope(const float* __restrict__ qkv,
                            const float* __restrict__ q_scale,
                            const float* __restrict__ k_scale,
                            bf16_t* __restrict__ Qb, bf16_t* __restrict__ Kb,
                            bf16_t* __restrict__ Vt) {
    const int wavesPerBlock = blockDim.x >> 5;
    int w = blockIdx.x * wavesPerBlock + (threadIdx.x >> 5);
    if (w >= B_ * L_ * H_) return;
    const int lane = threadIdx.x & 31;

    const int h = w % H_;
    const int l = (w / H_) % L_;
    const int b = w / (H_ * L_);

    const size_t rowbase = (size_t)(b * L_ + l) * TDIM;
    const size_t qkout   = ((size_t)(b * H_ + h) * L_ + l) * HD_;

    const float inv_freq = __expf(-((float)(2 * lane) / (float)HD_) * 9.210340371976184f);
    float sn, cs;
    __sincosf((float)l * inv_freq, &sn, &cs);

    // ---- Q ----
    {
        float x1 = qkv[rowbase + h * HD_ + lane];
        float x2 = qkv[rowbase + h * HD_ + lane + 32];
        float ss = x1 * x1 + x2 * x2;
#pragma unroll
        for (int o = 16; o >= 1; o >>= 1) ss += __shfl_xor(ss, o, 32);
        float inv = rsqrtf(ss * (1.0f / HD_) + 1e-6f);
        float n1 = x1 * inv * q_scale[lane];
        float n2 = x2 * inv * q_scale[lane + 32];
        Qb[qkout + lane]      = (bf16_t)(n1 * cs - n2 * sn);
        Qb[qkout + lane + 32] = (bf16_t)(n2 * cs + n1 * sn);
    }
    // ---- K ----
    {
        float x1 = qkv[rowbase + DIM_ + h * HD_ + lane];
        float x2 = qkv[rowbase + DIM_ + h * HD_ + lane + 32];
        float ss = x1 * x1 + x2 * x2;
#pragma unroll
        for (int o = 16; o >= 1; o >>= 1) ss += __shfl_xor(ss, o, 32);
        float inv = rsqrtf(ss * (1.0f / HD_) + 1e-6f);
        float n1 = x1 * inv * k_scale[lane];
        float n2 = x2 * inv * k_scale[lane + 32];
        Kb[qkout + lane]      = (bf16_t)(n1 * cs - n2 * sn);
        Kb[qkout + lane + 32] = (bf16_t)(n2 * cs + n1 * sn);
    }
    // ---- V (transpose to [hd][l]) ----
    {
        float v1 = qkv[rowbase + 2 * DIM_ + h * HD_ + lane];
        float v2 = qkv[rowbase + 2 * DIM_ + h * HD_ + lane + 32];
        const size_t vb = (size_t)(b * H_ + h) * HD_;
        Vt[(vb + lane)      * L_ + l] = (bf16_t)v1;
        Vt[(vb + lane + 32) * L_ + l] = (bf16_t)v2;
    }
}

// ---------------- flash attention ----------------
// blockDim = (32, 4): 4 waves share one (b,h); each owns one 16-query tile.
// K/V chunks (shared by all 4 waves) are streamed into double-buffered LDS with
// CDNA5 async global->LDS copies (ASYNCcnt), overlapping DMA with compute.

__global__ void k_attn(const bf16_t* __restrict__ Qb, const bf16_t* __restrict__ Kb,
                       const bf16_t* __restrict__ Vt, bf16_t* __restrict__ Ob) {
    const int lane = threadIdx.x;
    const int wid  = threadIdx.y;
    const int tid  = wid * 32 + lane;               // 0..127
    const int half = lane >> 4, l16 = lane & 15;
    const int qt = blockIdx.x * blockDim.y + wid;   // query tile 0..127
    const int bh = blockIdx.y;                      // 0..B*H-1
    const int b = bh / H_, h = bh % H_;

    __shared__ __align__(16) bf16_t Ksh[2][32][64];  // [buf][key_local][hd]
    __shared__ __align__(16) bf16_t Vsh[2][64][32];  // [buf][hd][key_local]
    __shared__ float   sc[4][16][32];
    __shared__ __align__(16) bf16_t pb[4][16][32];
    __shared__ float   salpha[4][16];
    __shared__ float   slinv[4][16];

    const bf16_t* Q = Qb + (size_t)bh * L_ * HD_;
    const bf16_t* K = Kb + (size_t)bh * L_ * HD_;
    const bf16_t* V = Vt + (size_t)bh * HD_ * L_;

    // Q fragments (HD=64 -> k0 = 0 and 32), register-resident for the whole loop
    const bf16_t* qrow = Q + (size_t)(qt * 16 + l16) * HD_;
    v16bf qf0 = make_frag(qrow + 0  + 8 * half, qrow + 16 + 8 * half);
    v16bf qf1 = make_frag(qrow + 32 + 8 * half, qrow + 48 + 8 * half);

    v8f acc[4];
#pragma unroll
    for (int t = 0; t < 4; ++t) acc[t] = (v8f){};
    float m_i = -3.0e38f, l_i = 0.0f;   // lanes 0..15 (row = lane)

    // issue async K/V loads for one 32-key chunk: 128 threads x 4 x 16B = 8KB
    auto issue_chunk = [&](int kk, int buf) {
#pragma unroll
        for (int r = 0; r < 2; ++r) {
            int u = tid + r * 128;                       // 0..255
            int krow = u >> 3, kcol = (u & 7) * 8;       // 32 rows x 64 bf16
            async_load16((unsigned)(size_t)&Ksh[buf][krow][kcol],
                         K + (size_t)(kk + krow) * HD_ + kcol);
            int vrow = u >> 2, vcol = (u & 3) * 8;       // 64 rows x 32 bf16
            async_load16((unsigned)(size_t)&Vsh[buf][vrow][vcol],
                         V + (size_t)vrow * L_ + kk + vcol);
        }
    };

    issue_chunk(0, 0);

    const int NCHUNK = L_ / 32;
    for (int c = 0; c < NCHUNK; ++c) {
        const int buf = c & 1;
        wait_async0();        // my chunk-c copies complete
        __syncthreads();      // everyone's copies visible; previous compute done
        if (c + 1 < NCHUNK) issue_chunk((c + 1) * 32, buf ^ 1);

        // ---- S = Q @ K_chunk^T (16x32), K from LDS ----
#pragma unroll
        for (int nh = 0; nh < 2; ++nh) {
            v8f sacc = (v8f){};
            const bf16_t* krow = &Ksh[buf][nh * 16 + l16][0];
            v16bf kf0 = make_frag(krow + 0  + 8 * half, krow + 16 + 8 * half);
            v16bf kf1 = make_frag(krow + 32 + 8 * half, krow + 48 + 8 * half);
            sacc = __builtin_amdgcn_wmma_f32_16x16x32_bf16(false, qf0, false, kf0,
                                                           (short)0, sacc, false, false);
            sacc = __builtin_amdgcn_wmma_f32_16x16x32_bf16(false, qf1, false, kf1,
                                                           (short)0, sacc, false, false);
#pragma unroll
            for (int v = 0; v < 8; ++v)
                sc[wid][v + 8 * half][nh * 16 + l16] = sacc[v];
        }

        // ---- online softmax (lanes 0..15, row = lane) ----
        if (lane < 16) {
            float sv[32];
            float mx = m_i;
#pragma unroll
            for (int j = 0; j < 32; ++j) {
                sv[j] = sc[wid][lane][j] * 0.125f;   // HD^-0.5
                mx = fmaxf(mx, sv[j]);
            }
            float sum = 0.0f;
#pragma unroll
            for (int j = 0; j < 32; ++j) {
                float p = __expf(sv[j] - mx);
                sum += p;
                pb[wid][lane][j] = (bf16_t)p;
            }
            float alpha = __expf(m_i - mx);
            l_i = l_i * alpha + sum;
            m_i = mx;
            salpha[wid][lane] = alpha;
        }

        // ---- rescale running O accumulators ----
#pragma unroll
        for (int v = 0; v < 8; ++v) {
            float a = salpha[wid][v + 8 * half];
#pragma unroll
            for (int t = 0; t < 4; ++t) acc[t][v] *= a;
        }

        // ---- O += P @ V_chunk, V from LDS ----
        const bf16_t* prow = &pb[wid][l16][0];
        v16bf pf = make_frag(prow + 16 * half, prow + 16 * half + 8);
#pragma unroll
        for (int t = 0; t < 4; ++t) {
            const bf16_t* vrow = &Vsh[buf][16 * t + l16][16 * half];
            v16bf vf = make_frag(vrow, vrow + 8);
            acc[t] = __builtin_amdgcn_wmma_f32_16x16x32_bf16(false, pf, false, vf,
                                                             (short)0, acc[t], false, false);
        }
    }

    if (lane < 16) slinv[wid][lane] = 1.0f / l_i;

    // ---- write O (bf16, layout [b][l][h][hd] == GEMM2 A [B*L][DIM]) ----
#pragma unroll
    for (int t = 0; t < 4; ++t)
#pragma unroll
        for (int v = 0; v < 8; ++v) {
            int row = qt * 16 + v + 8 * half;
            float o = acc[t][v] * slinv[wid][v + 8 * half];
            Ob[((size_t)(b * L_ + row) * H_ + h) * HD_ + 16 * t + l16] = (bf16_t)o;
        }
}

// ---------------- host launcher ----------------

extern "C" void kernel_launch(void* const* d_in, const int* in_sizes, int n_in,
                              void* d_out, int out_size, void* d_ws, size_t ws_size,
                              hipStream_t stream) {
    const float* x       = (const float*)d_in[0];
    const float* Wqkv    = (const float*)d_in[1];
    const float* q_scale = (const float*)d_in[2];
    const float* k_scale = (const float*)d_in[3];
    const float* Wproj   = (const float*)d_in[4];
    const float* bproj   = (const float*)d_in[5];
    float* out = (float*)d_out;

    char* ws = (char*)d_ws;
    size_t off = 0;
    auto alloc = [&](size_t bytes) -> void* {
        void* p = ws + off;
        off += (bytes + 255) & ~(size_t)255;
        return p;
    };

    const size_t ML = (size_t)B_ * L_;           // 4096 rows
    float*  qkv    = (float*) alloc(ML * TDIM * sizeof(float));
    bf16_t* xb     = (bf16_t*)alloc(ML * DIM_ * sizeof(bf16_t));
    bf16_t* wqkvT  = (bf16_t*)alloc((size_t)TDIM * DIM_ * sizeof(bf16_t));
    bf16_t* wprojT = (bf16_t*)alloc((size_t)DIM_ * DIM_ * sizeof(bf16_t));
    bf16_t* Qb     = (bf16_t*)alloc(ML * DIM_ * sizeof(bf16_t));
    bf16_t* Kb     = (bf16_t*)alloc(ML * DIM_ * sizeof(bf16_t));
    bf16_t* Vt     = (bf16_t*)alloc(ML * DIM_ * sizeof(bf16_t));
    bf16_t* Ob     = (bf16_t*)alloc(ML * DIM_ * sizeof(bf16_t));

    // 1) convert inputs
    {
        int n = (int)(ML * DIM_);
        k_cvt_f32_bf16<<<(n + 255) / 256, 256, 0, stream>>>(x, xb, n);
        int nq = DIM_ * TDIM;
        k_transpose_cvt<<<(nq + 255) / 256, 256, 0, stream>>>(Wqkv, wqkvT, DIM_, TDIM);
        int np = DIM_ * DIM_;
        k_transpose_cvt<<<(np + 255) / 256, 256, 0, stream>>>(Wproj, wprojT, DIM_, DIM_);
    }

    // 2) QKV GEMM: [4096,1024] @ [1024,3072] -> fp32
    k_gemm_wmma<<<dim3(ML / 32, TDIM / 64), 32, 0, stream>>>(
        xb, wqkvT, qkv, nullptr, (int)ML, TDIM, DIM_);

    // 3) RMSNorm + RoPE + pack Q/K/V
    {
        int rows = B_ * L_ * H_;             // 65536 waves
        k_norm_rope<<<rows / 8, 256, 0, stream>>>(qkv, q_scale, k_scale, Qb, Kb, Vt);
    }

    // 4) flash attention: 128 query tiles x 32 heads
    k_attn<<<dim3(L_ / 16 / 4, B_ * H_), dim3(32, 4), 0, stream>>>(Qb, Kb, Vt, Ob);

    // 5) output projection + bias -> fp32 d_out
    k_gemm_wmma<<<dim3(ML / 32, DIM_ / 64), 32, 0, stream>>>(
        Ob, wprojT, out, bproj, (int)ML, DIM_, DIM_);

    (void)in_sizes; (void)n_in; (void)out_size; (void)ws_size;
}
